// BNN_34127810134269
// MI455X (gfx1250) — matrix operand order repaired and checked by reading, not compiled, not run on hardware
//
#include <hip/hip_runtime.h>
#include <stdint.h>

// ---------------------------------------------------------------------------
// Bayesian MLP MC-sampling, fused: W = eps*exp(0.5*lv)+mean generated on the
// fly into bf16 WMMA B-fragments; activations live in LDS; eps streamed NT.
// HBM-bound (~270MB eps stream @ 23.3 TB/s => ~12us floor). This revision
// batches each fragment's 48 independent loads into one deep clause (register
// arrays + sched_barrier that blocks VMEM motion only), replacing the
// load->wait(0)->use serialization the scheduler produced in round 2.
// ---------------------------------------------------------------------------

#define NSAMP 100
#define BATCH 64
#define D0 784
#define D1 512
#define D2 512
#define D3 10
#define K0PAD 800   // 784 padded to multiple of 32 (LDS A-buffer only)

typedef __attribute__((ext_vector_type(16))) __bf16 v16bf;
typedef __attribute__((ext_vector_type(8)))  float  v8f;

union Frag16 {
    v16bf    bf;
    __bf16   h[16];
    uint32_t u[8];
};

__device__ __forceinline__ uint16_t to_bf16_bits(float f) {
    return __builtin_bit_cast(uint16_t, (__bf16)f);   // native v_cvt bf16
}

// exp(0.5*x) = 2^(x * 0.5*log2(e))
#define HALF_LOG2E 0.72134752044448170f

// sched_barrier mask: allow ALU/VALU/SALU/WMMA/DS/trans to cross, block VMEM.
// 0x1 all-non-mem | 0x2 VALU | 0x4 SALU | 0x8 MFMA/WMMA | 0x80/0x100/0x200 DS | 0x400 trans
#define SCHED_NO_VMEM_CROSS 0x078F
__device__ __forceinline__ void fence_vmem_sched() {
#if __has_builtin(__builtin_amdgcn_sched_barrier)
    __builtin_amdgcn_sched_barrier(SCHED_NO_VMEM_CROSS);
#endif
}

// Build one 32(K)x16(N) sampled-weight B fragment.
// B k order used here (contiguous, as load_A): lanes 0-15 hold K=0..15 (2 per VGPR, K=2j,2j+1),
// lanes 16-31 hold K=16..31.  TAIL: branchless clamp+mask for K overhang.
// Phase 1 issues all 48 loads (one deep clause); phase 2 transforms/packs.
template <bool TAIL>
__device__ __forceinline__ Frag16 sample_B(
    const float* __restrict__ eps, const float* __restrict__ lv,
    const float* __restrict__ mn, int k0, int bhalf, int n, int kdim)
{
    float ev[16], lw[16], mw[16];
#pragma unroll
    for (int j = 0; j < 8; ++j) {
        int k = k0 + bhalf + 2 * j;
        if (TAIL) k = (k < kdim) ? k : 0;            // clamp: loads stay in-bounds
        const size_t i0 = (size_t)k * 512 + n;
        ev[2 * j]     = __builtin_nontemporal_load(eps + i0);
        ev[2 * j + 1] = __builtin_nontemporal_load(eps + i0 + 512);
        lw[2 * j]     = lv[i0];
        lw[2 * j + 1] = lv[i0 + 512];
        mw[2 * j]     = mn[i0];
        mw[2 * j + 1] = mn[i0 + 512];
    }
    fence_vmem_sched();                               // keep loads batched above

    Frag16 B;
#pragma unroll
    for (int j = 0; j < 8; ++j) {
        float msk = 1.f;
        if (TAIL) {
            const int k = k0 + bhalf + 2 * j;
            msk = (k < kdim) ? 1.f : 0.f;             // v_cndmask, EXEC all-ones
        }
        float w0 = fmaf(ev[2 * j],     __builtin_amdgcn_exp2f(HALF_LOG2E * lw[2 * j]),     mw[2 * j]);
        float w1 = fmaf(ev[2 * j + 1], __builtin_amdgcn_exp2f(HALF_LOG2E * lw[2 * j + 1]), mw[2 * j + 1]);
        if (TAIL) { w0 *= msk; w1 *= msk; }
        B.h[2 * j]     = (__bf16)w0;                  // -> v_cvt_pk_bf16_f32
        B.h[2 * j + 1] = (__bf16)w1;
    }
    return B;
}

// A k order used here (contiguous): lane-half -> K+16, VGPR group (j>=4) -> K+8,
// so per lane the fragment is two 8x-bf16 runs -> 2x ds_load_b128.
__device__ __forceinline__ Frag16 load_A(const uint16_t* __restrict__ Ain,
                                         int kpad, int row, int k0, int ahalf)
{
    Frag16 A;
    const uint16_t* p = Ain + (size_t)row * kpad + k0 + ahalf;
    const uint4 g0 = *(const uint4*)(p);
    const uint4 g1 = *(const uint4*)(p + 8);
    A.u[0] = g0.x; A.u[1] = g0.y; A.u[2] = g0.z; A.u[3] = g0.w;
    A.u[4] = g1.x; A.u[5] = g1.y; A.u[6] = g1.z; A.u[7] = g1.w;
    return A;
}

// One hidden layer: out[32,512] = relu(in[32,K] @ W[K,512] + b), W,b sampled.
// 8 waves x 64-wide N panels; acc[2 Mtiles][4 Ntiles] per wave.
template <int KDIM, int KPAD>
__device__ __forceinline__ void hidden_layer(
    const uint16_t* __restrict__ Ain,      // LDS bf16 [32][KPAD]
    const float* __restrict__ eps,         // [KDIM][512] (this sample)
    const float* __restrict__ lv,
    const float* __restrict__ mn,
    const float* __restrict__ beps,        // [512] (this sample)
    const float* __restrict__ blv,
    const float* __restrict__ bmn,
    uint16_t* __restrict__ Aout,           // LDS bf16 [32][512]
    int lane, int wave)
{
    v8f acc[2][4] = {};
    const int nb    = wave * 64;
    const int mrow  = lane & 15;
    const int ahalf = (lane >> 4) << 4;
    const int bhalf = (lane >> 4) << 4;
    const int ncol  = lane & 15;

    constexpr int KMAIN = KDIM & ~31;      // guard-free chunks

    for (int k0 = 0; k0 < KMAIN; k0 += 32) {
        Frag16 A[2];
#pragma unroll
        for (int mt = 0; mt < 2; ++mt)
            A[mt] = load_A(Ain, KPAD, mt * 16 + mrow, k0, ahalf);
#pragma unroll
        for (int nt = 0; nt < 4; ++nt) {
            const Frag16 B = sample_B<false>(eps, lv, mn, k0, bhalf, nb + nt * 16 + ncol, KDIM);
#pragma unroll
            for (int mt = 0; mt < 2; ++mt)
                acc[mt][nt] = __builtin_amdgcn_wmma_f32_16x16x32_bf16(
                    false, A[mt].bf, false, B.bf, (short)0, acc[mt][nt], false, false);
        }
    }
    if constexpr (KDIM % 32 != 0) {        // single branchless tail chunk
        Frag16 A[2];
#pragma unroll
        for (int mt = 0; mt < 2; ++mt)
            A[mt] = load_A(Ain, KPAD, mt * 16 + mrow, KMAIN, ahalf);  // LDS zero-padded
#pragma unroll
        for (int nt = 0; nt < 4; ++nt) {
            const Frag16 B = sample_B<true>(eps, lv, mn, KMAIN, bhalf, nb + nt * 16 + ncol, KDIM);
#pragma unroll
            for (int mt = 0; mt < 2; ++mt)
                acc[mt][nt] = __builtin_amdgcn_wmma_f32_16x16x32_bf16(
                    false, A[mt].bf, false, B.bf, (short)0, acc[mt][nt], false, false);
        }
    }

    // epilogue: sampled bias, relu, bf16 -> LDS
#pragma unroll
    for (int nt = 0; nt < 4; ++nt) {
        const int n = nb + nt * 16 + ncol;
        const float bias = fmaf(beps[n], __builtin_amdgcn_exp2f(HALF_LOG2E * blv[n]), bmn[n]);
#pragma unroll
        for (int mt = 0; mt < 2; ++mt) {
#pragma unroll
            for (int v = 0; v < 8; ++v) {
                const int r = mt * 16 + v + ((lane >> 4) << 3);  // C/D layout
                float val = acc[mt][nt][v] + bias;
                val = val > 0.f ? val : 0.f;
                Aout[(size_t)r * 512 + n] = to_bf16_bits(val);
            }
        }
    }
}

__global__ __launch_bounds__(256) void bnn_mc_kernel(
    const float* __restrict__ inputs,
    const float* __restrict__ wm0, const float* __restrict__ wv0,
    const float* __restrict__ bm0, const float* __restrict__ bv0,
    const float* __restrict__ wm1, const float* __restrict__ wv1,
    const float* __restrict__ bm1, const float* __restrict__ bv1,
    const float* __restrict__ wml, const float* __restrict__ wvl,
    const float* __restrict__ bml, const float* __restrict__ bvl,
    const float* __restrict__ we0, const float* __restrict__ be0,
    const float* __restrict__ we1, const float* __restrict__ be1,
    const float* __restrict__ wel, const float* __restrict__ bel,
    float* __restrict__ out)
{
    extern __shared__ __align__(16) char smem[];
    uint16_t* bufA = (uint16_t*)smem;                          // [32][800], reused [32][512]
    uint16_t* bufB = (uint16_t*)(smem + 32 * K0PAD * 2);       // [32][512]

    const int s     = blockIdx.x >> 1;         // sample
    const int rbase = (blockIdx.x & 1) * 32;   // batch-row half (pair shares eps via L2)
    const int tid   = threadIdx.x;
    const int lane  = tid & 31;
    const int wave  = tid >> 5;

    // Stage input rows [rbase, rbase+32) as bf16, zero-pad cols 784..799.
    for (int idx = tid; idx < 32 * K0PAD; idx += 256) {
        const int r = idx / K0PAD;
        const int c = idx - r * K0PAD;
        const float v = (c < D0) ? inputs[(size_t)(rbase + r) * D0 + c] : 0.f;
        bufA[idx] = to_bf16_bits(v);
    }
    __syncthreads();

    // Layer 0: bufA[32,800] -> bufB[32,512]
    hidden_layer<D0, K0PAD>(bufA,
        we0 + (size_t)s * D0 * D1, wv0, wm0,
        be0 + (size_t)s * D1, bv0, bm0,
        bufB, lane, wave);
    __syncthreads();

    // Layer 1: bufB[32,512] -> bufA[32,512] (reused)
    hidden_layer<D1, D1>(bufB,
        we1 + (size_t)s * D1 * D2, wv1, wm1,
        be1 + (size_t)s * D2, bv1, bm1,
        bufA, lane, wave);
    __syncthreads();

    // Layer 2: logits [32,10] = one 16-wide N tile; waves 0..1 take M tiles.
    // N<16 handled branchlessly: clamp column, mask sampled weights to zero.
    if (wave < 2) {
        const int mt    = wave;
        const int mrow  = lane & 15;
        const int ahalf = (lane >> 4) << 4;
        const int bhalf = (lane >> 4) << 4;
        const int n     = lane & 15;
        const int   nc    = (n < D3) ? n   : (D3 - 1);   // in-bounds column
        const float nmask = (n < D3) ? 1.f : 0.f;
        const float* eps = wel + (size_t)s * D2 * D3;
        v8f acc = {};
        for (int k0 = 0; k0 < D2; k0 += 32) {
            const Frag16 A = load_A(bufA, 512, mt * 16 + mrow, k0, ahalf);
            float ev[16], lw[16], mw[16];
#pragma unroll
            for (int j = 0; j < 8; ++j) {              // phase 1: batched loads
                const size_t i0 = (size_t)(k0 + bhalf + 2 * j) * D3 + nc;
                ev[2 * j]     = __builtin_nontemporal_load(eps + i0);
                ev[2 * j + 1] = __builtin_nontemporal_load(eps + i0 + D3);
                lw[2 * j]     = wvl[i0];
                lw[2 * j + 1] = wvl[i0 + D3];
                mw[2 * j]     = wml[i0];
                mw[2 * j + 1] = wml[i0 + D3];
            }
            fence_vmem_sched();
            Frag16 B;
#pragma unroll
            for (int j = 0; j < 8; ++j) {              // phase 2: transform/pack
                const float w0 = nmask * fmaf(ev[2 * j],
                    __builtin_amdgcn_exp2f(HALF_LOG2E * lw[2 * j]), mw[2 * j]);
                const float w1 = nmask * fmaf(ev[2 * j + 1],
                    __builtin_amdgcn_exp2f(HALF_LOG2E * lw[2 * j + 1]), mw[2 * j + 1]);
                B.h[2 * j]     = (__bf16)w0;
                B.h[2 * j + 1] = (__bf16)w1;
            }
            acc = __builtin_amdgcn_wmma_f32_16x16x32_bf16(
                false, A.bf, false, B.bf, (short)0, acc, false, false);
        }
        if (n < D3) {
            const float bias = fmaf(bel[(size_t)s * D3 + n],
                                    __builtin_amdgcn_exp2f(HALF_LOG2E * bvl[n]), bml[n]);
#pragma unroll
            for (int v = 0; v < 8; ++v) {
                const int r = rbase + mt * 16 + v + ((lane >> 4) << 3);
                out[(size_t)s * BATCH * D3 + (size_t)r * D3 + n] = acc[v] + bias;
            }
        }
    }
}

extern "C" void kernel_launch(void* const* d_in, const int* in_sizes, int n_in,
                              void* d_out, int out_size, void* d_ws, size_t ws_size,
                              hipStream_t stream) {
    (void)in_sizes; (void)n_in; (void)d_ws; (void)ws_size; (void)out_size;
    const float* inputs = (const float*)d_in[0];
    // d_in[1] = task_id (unused: single head)
    const float* wm0 = (const float*)d_in[2];
    const float* wv0 = (const float*)d_in[3];
    const float* bm0 = (const float*)d_in[4];
    const float* bv0 = (const float*)d_in[5];
    const float* wm1 = (const float*)d_in[6];
    const float* wv1 = (const float*)d_in[7];
    const float* bm1 = (const float*)d_in[8];
    const float* bv1 = (const float*)d_in[9];
    const float* wml = (const float*)d_in[10];
    const float* wvl = (const float*)d_in[11];
    const float* bml = (const float*)d_in[12];
    const float* bvl = (const float*)d_in[13];
    const float* we0 = (const float*)d_in[14];
    const float* be0 = (const float*)d_in[15];
    const float* we1 = (const float*)d_in[16];
    const float* be1 = (const float*)d_in[17];
    const float* wel = (const float*)d_in[18];
    const float* bel = (const float*)d_in[19];
    float* out = (float*)d_out;

    const size_t smem_bytes = (size_t)32 * K0PAD * 2   // bufA: 51,200 B
                            + (size_t)32 * 512 * 2;    // bufB: 32,768 B -> 84 KB < 320 KB WGP LDS
    bnn_mc_kernel<<<dim3(2 * NSAMP), dim3(256), smem_bytes, stream>>>(
        inputs, wm0, wv0, bm0, bv0, wm1, wv1, bm1, bv1,
        wml, wvl, bml, bvl, we0, be0, we1, be1, wel, bel, out);
}
